// VisualizeInternalGates_24927990186539
// MI455X (gfx1250) — compile-verified, matrix operand order checked by reading, not run on hardware
//
#include <hip/hip_runtime.h>

typedef unsigned short u16;
typedef __attribute__((ext_vector_type(16))) unsigned short v16u;
typedef __attribute__((ext_vector_type(16))) __bf16         v16bf;
typedef __attribute__((ext_vector_type(8)))  float          v8f;

constexpr int NB = 256;   // batch
constexpr int NT = 512;   // time
constexpr int NV = 82;    // vocab
constexpr int NE = 256;   // embedding dim
constexpr int NH = 128;   // hidden
constexpr int LDW3 = 2 * NH + NE;  // 512, ld of W_f/W_i/W_o
constexpr int LDWC = NH + NE;      // 384, ld of W_c
constexpr int LDH  = 136;          // padded LDS row stride (u16): 272B = 68 dwords -> 4-bank shift/row
constexpr size_t TBH = (size_t)NT * NB * NH;

// ---------- small helpers ----------

__device__ inline u16 f2bf(float f) {
  union { float f; unsigned u; } a; a.f = f;
  unsigned u = a.u;
  unsigned r = u + 0x7FFFu + ((u >> 16) & 1u);  // round-to-nearest-even
  return (u16)(r >> 16);
}

#if __has_builtin(__builtin_amdgcn_tanhf)
__device__ inline float fast_tanh(float x) { return __builtin_amdgcn_tanhf(x); }
#elif __has_builtin(__builtin_amdgcn_tanh_f32)
__device__ inline float fast_tanh(float x) { return __builtin_amdgcn_tanh_f32(x); }
#else
__device__ inline float fast_tanh(float x) { return tanhf(x); }
#endif

__device__ inline float fast_sigmoid(float x) {
  return 0.5f * fast_tanh(0.5f * x) + 0.5f;   // one TRANS op on gfx1250
}

__device__ inline v16bf as_bf(v16u v) { return __builtin_bit_cast(v16bf, v); }

__device__ inline v8f wmma_bf(v16bf a, v16bf b, v8f c) {
  // D = A(16x32 bf16) * B(32x16 bf16) + C(16x16 f32)
  return __builtin_amdgcn_wmma_f32_16x16x32_bf16(false, a, false, b, (short)0, c,
                                                 false, false);
}

// A/B fragment from row-major bf16 source (global or LDS): rows contiguous along K.
// Lane l<16 holds row (row0+l), K = k0+{0..7, 16..23}; lanes 16..31 K = k0+{8..15, 24..31}.
__device__ inline v16bf load_ab_bf16(const u16* src, int row0, int k0, int ld, int lane) {
  int r  = lane & 15;
  int kb = k0 + ((lane & 16) ? 8 : 0);
  const u16* p = src + (size_t)(row0 + r) * ld + kb;
  v16u v;
#pragma unroll
  for (int j = 0; j < 8; ++j) { v[j] = p[j]; v[8 + j] = p[16 + j]; }
  return as_bf(v);
}

// Row-guarded variant (for W_cls where N=82 is not a multiple of 16).
__device__ inline v16bf load_ab_bf16_guard(const u16* src, int row0, int k0, int ld,
                                           int nrows, int lane) {
  int r   = lane & 15;
  int row = row0 + r;
  v16u v;
  if (row < nrows) {
    int kb = k0 + ((lane & 16) ? 8 : 0);
    const u16* p = src + (size_t)row * ld + kb;
#pragma unroll
    for (int j = 0; j < 8; ++j) { v[j] = p[j]; v[8 + j] = p[16 + j]; }
  } else {
#pragma unroll
    for (int j = 0; j < 16; ++j) v[j] = 0;
  }
  return as_bf(v);
}

// C/D fragment <-> row-major f32: vgpr j holds m = m0 + j + 8*(lane>=16), n = n0 + (lane&15)
__device__ inline v8f load_cd_f32(const float* src, int m0, int n0, int ld, int lane) {
  int n  = n0 + (lane & 15);
  int mb = m0 + ((lane & 16) ? 8 : 0);
  v8f v;
#pragma unroll
  for (int j = 0; j < 8; ++j) v[j] = src[(size_t)(mb + j) * ld + n];
  return v;
}

__device__ inline void store_cd_f32(float* dst, int m0, int n0, int ld, int lane, v8f v) {
  int n  = n0 + (lane & 15);
  int mb = m0 + ((lane & 16) ? 8 : 0);
#pragma unroll
  for (int j = 0; j < 8; ++j) dst[(size_t)(mb + j) * ld + n] = v[j];
}

__device__ inline void store_cd_bf16(u16* dst, int m0, int n0, int ld, int lane, v8f v) {
  int n  = n0 + (lane & 15);
  int mb = m0 + ((lane & 16) ? 8 : 0);
#pragma unroll
  for (int j = 0; j < 8; ++j) dst[(size_t)(mb + j) * ld + n] = f2bf(v[j]);
}

// ---------- kernel 0: convert weights/embedding to bf16 ----------

__global__ void prep_kernel(const float* emb, const float* W_f, const float* W_i,
                            const float* W_o, const float* W_c, const float* W_cls,
                            u16* embbf, u16* wf, u16* wi, u16* wo, u16* wc, u16* wcls) {
  int idx = blockIdx.x * blockDim.x + threadIdx.x;
  int stride = gridDim.x * blockDim.x;
  for (int i = idx; i < NV * NE; i += stride) embbf[i] = f2bf(emb[i]);
  for (int i = idx; i < NH * LDW3; i += stride) {
    wf[i] = f2bf(W_f[i]); wi[i] = f2bf(W_i[i]); wo[i] = f2bf(W_o[i]);
  }
  for (int i = idx; i < NH * LDWC; i += stride) wc[i] = f2bf(W_c[i]);
  for (int i = idx; i < NV * NH; i += stride) wcls[i] = f2bf(W_cls[i]);
}

// ---------- kernel 1: precompute x-projections for all (t,b), all 4 gates ----------
// Px_g[M, :] = emb[tok(M)] @ Wg_x.T + b_g, M = t*NB + b. f/i/o staged in d_out, c in ws.

__global__ __launch_bounds__(256)
void precompute_kernel(const int* tokens, const u16* embbf,
                       const u16* wf, const u16* wi, const u16* wo, const u16* wc,
                       const float* b_f, const float* b_i, const float* b_o, const float* b_c,
                       float* out_f, float* out_i, float* out_o, float* px_c) {
  int lane = threadIdx.x & 31;
  int nt   = threadIdx.x >> 5;       // 8 waves -> 8 n-tiles (H=128)
  int m0   = blockIdx.x * 16;        // 8192 m-tiles over T*B
  int g    = blockIdx.y;             // gate

  const u16* W; const float* bias; float* dst; int ldw, xoff;
  if      (g == 0) { W = wf; bias = b_f; dst = out_f; ldw = LDW3; xoff = 2 * NH; }
  else if (g == 1) { W = wi; bias = b_i; dst = out_i; ldw = LDW3; xoff = 2 * NH; }
  else if (g == 2) { W = wo; bias = b_o; dst = out_o; ldw = LDW3; xoff = 2 * NH; }
  else             { W = wc; bias = b_c; dst = px_c; ldw = LDWC; xoff = NH; }

  // per-lane token gather for the A fragment row
  int M   = m0 + (lane & 15);
  int t   = M / NB;
  int b   = M & (NB - 1);
  int tok = tokens[b * NT + t];
  const u16* arow = embbf + (size_t)tok * NE;

  int n0 = nt * 16;
  v8f acc;
#pragma unroll
  for (int j = 0; j < 8; ++j) acc[j] = 0.f;

#pragma unroll
  for (int kc = 0; kc < NE / 32; ++kc) {
    int k0 = kc * 32;
    int kb = k0 + ((lane & 16) ? 8 : 0);
    v16u av;
#pragma unroll
    for (int j = 0; j < 8; ++j) { av[j] = arow[kb + j]; av[8 + j] = arow[kb + 16 + j]; }
    v16bf afrag = as_bf(av);
    v16bf bfrag = load_ab_bf16(W + xoff, n0, k0, ldw, lane);
    acc = wmma_bf(afrag, bfrag, acc);
  }

  float bb = bias[n0 + (lane & 15)];
#pragma unroll
  for (int j = 0; j < 8; ++j) acc[j] += bb;

  store_cd_f32(dst, m0, n0, NH, lane, acc);
}

// ---------- kernel 2: LSTM recurrence, batch-parallel across 16 WGPs ----------
// Batch rows are independent: workgroup g owns rows [16g, 16g+16). State h/c (bf16)
// ping-pongs in LDS; the f32 cell state lives in each wave's registers (wave owns a
// fixed 16x16 tile for the whole kernel). h-part weight fragments preloaded in VGPRs.

__global__ __launch_bounds__(256, 1)
void recurrent_kernel(const u16* wf, const u16* wi, const u16* wo, const u16* wc,
                      float* out_f, float* out_i, float* out_o, const float* px_c,
                      u16* hfin) {
  __shared__ u16 sh_h[2][16 * LDH];
  __shared__ u16 sh_c[2][16 * LDH];

  int lane = threadIdx.x & 31;
  int nt   = threadIdx.x >> 5;        // 8 waves -> 8 n-tiles
  int tid  = threadIdx.x;
  int m0g  = blockIdx.x * 16;         // this workgroup's batch rows
  int n0   = nt * 16;

  // zero initial state (h0 = c0 = 0)
  for (int i = tid; i < 2 * 16 * LDH; i += 256) {
    (&sh_h[0][0])[i] = 0;
    (&sh_c[0][0])[i] = 0;
  }

  // preload h-part B fragments for this wave's fixed n-tile (reused all 512 steps)
  v16bf Bfh[4], Bih[4], Boh[4], Bch[4];
#pragma unroll
  for (int kc = 0; kc < 4; ++kc) {
    int k0 = kc * 32;
    Bfh[kc] = load_ab_bf16(wf + NH, n0, k0, LDW3, lane);
    Bih[kc] = load_ab_bf16(wi + NH, n0, k0, LDW3, lane);
    Boh[kc] = load_ab_bf16(wo + NH, n0, k0, LDW3, lane);
    Bch[kc] = load_ab_bf16(wc,      n0, k0, LDWC, lane);  // W_c h-part is cols [0,H)
  }

  v8f c_reg;                          // f32 cell state, persistent in registers
#pragma unroll
  for (int j = 0; j < 8; ++j) c_reg[j] = 0.f;

  v8f hn;
#pragma unroll
  for (int j = 0; j < 8; ++j) hn[j] = 0.f;

  __syncthreads();

  for (int t = 0; t < NT; ++t) {
    int cur = t & 1, nxt = cur ^ 1;
    float* pf = out_f + (size_t)t * NB * NH;
    float* pi = out_i + (size_t)t * NB * NH;
    float* po = out_o + (size_t)t * NB * NH;
    const float* pc = px_c + (size_t)t * NB * NH;

    // prefetch next step's Px rows for this workgroup (4 arrays x 32KB / 256 thr)
    if (t + 1 < NT) {
      size_t off = (size_t)(t + 1) * NB * NH + (size_t)m0g * NH + (size_t)tid * 32;
      __builtin_prefetch(out_f + off, 0, 1);
      __builtin_prefetch(out_i + off, 0, 1);
      __builtin_prefetch(out_o + off, 0, 1);
      __builtin_prefetch(px_c + off, 0, 1);
    }

    // start from the precomputed x-projection (+bias)
    v8f af = load_cd_f32(pf, m0g, n0, NH, lane);
    v8f ai = load_cd_f32(pi, m0g, n0, NH, lane);
    v8f ao = load_cd_f32(po, m0g, n0, NH, lane);
    v8f ac = load_cd_f32(pc, m0g, n0, NH, lane);

#pragma unroll
    for (int kc = 0; kc < 4; ++kc) {
      int k0 = kc * 32;
      v16bf ah = load_ab_bf16(&sh_h[cur][0], 0, k0, LDH, lane);   // ds_load
      v16bf av = load_ab_bf16(&sh_c[cur][0], 0, k0, LDH, lane);   // ds_load
      af = wmma_bf(av, load_ab_bf16(wf, n0, k0, LDW3, lane), af); // c-part (WGP$)
      af = wmma_bf(ah, Bfh[kc], af);                              // h-part (regs)
      ai = wmma_bf(av, load_ab_bf16(wi, n0, k0, LDW3, lane), ai);
      ai = wmma_bf(ah, Bih[kc], ai);
      ao = wmma_bf(av, load_ab_bf16(wo, n0, k0, LDW3, lane), ao);
      ao = wmma_bf(ah, Boh[kc], ao);
      ac = wmma_bf(ah, Bch[kc], ac);
    }

    v8f fv, iv, ov;
#pragma unroll
    for (int j = 0; j < 8; ++j) {
      float f  = fast_sigmoid(af[j]);
      float i  = fast_sigmoid(ai[j]);
      float o  = fast_sigmoid(ao[j]);
      float ct = fast_tanh(ac[j]);
      float nc = f * c_reg[j] + i * ct;
      c_reg[j] = nc;
      fv[j] = f; iv[j] = i; ov[j] = o;
      hn[j] = o * fast_tanh(nc);
    }

    store_cd_f32(pf, m0g, n0, NH, lane, fv);
    store_cd_f32(pi, m0g, n0, NH, lane, iv);
    store_cd_f32(po, m0g, n0, NH, lane, ov);
    store_cd_bf16(&sh_c[nxt][0], 0, n0, LDH, lane, c_reg);
    store_cd_bf16(&sh_h[nxt][0], 0, n0, LDH, lane, hn);

    __syncthreads();
  }

  // final h of this wave's tile is still in registers
  store_cd_bf16(hfin, m0g, n0, NH, lane, hn);
}

// ---------- kernel 3: classifier logits = h_final @ W_cls.T + b_cls ----------

__global__ __launch_bounds__(192)
void classifier_kernel(const u16* hfin, const u16* wcls, const float* bcls,
                       float* logits) {
  int lane = threadIdx.x & 31;
  int nt   = threadIdx.x >> 5;   // 6 waves -> 6 n-tiles (ceil(82/16))
  int m0   = blockIdx.x * 16;    // 16 m-tiles over B=256
  int n0   = nt * 16;

  v8f acc;
#pragma unroll
  for (int j = 0; j < 8; ++j) acc[j] = 0.f;

#pragma unroll
  for (int kc = 0; kc < NH / 32; ++kc) {
    v16bf a = load_ab_bf16(hfin, m0, kc * 32, NH, lane);
    v16bf b = load_ab_bf16_guard(wcls, n0, kc * 32, NH, NV, lane);
    acc = wmma_bf(a, b, acc);
  }

  int n = n0 + (lane & 15);
  if (n < NV) {
    float bb = bcls[n];
    int mb = m0 + ((lane & 16) ? 8 : 0);
#pragma unroll
    for (int j = 0; j < 8; ++j) logits[(size_t)(mb + j) * NV + n] = acc[j] + bb;
  }
}

// ---------- launch ----------

extern "C" void kernel_launch(void* const* d_in, const int* in_sizes, int n_in,
                              void* d_out, int out_size, void* d_ws, size_t ws_size,
                              hipStream_t stream) {
  const int*   tokens = (const int*)  d_in[0];
  const float* emb    = (const float*)d_in[1];
  const float* W_f    = (const float*)d_in[2];
  const float* W_i    = (const float*)d_in[3];
  const float* W_o    = (const float*)d_in[4];
  const float* W_c    = (const float*)d_in[5];
  const float* b_f    = (const float*)d_in[6];
  const float* b_i    = (const float*)d_in[7];
  const float* b_o    = (const float*)d_in[8];
  const float* b_c    = (const float*)d_in[9];
  const float* W_cls  = (const float*)d_in[10];
  const float* b_cls  = (const float*)d_in[11];

  float* out    = (float*)d_out;
  float* logits = out;                  // [B, V]
  float* out_f  = out + NB * NV;        // [T, B, H]
  float* out_i  = out_f + TBH;
  float* out_o  = out_i + TBH;

  // workspace layout: bf16 block (weights + emb + final-h) | Px_c f32
  u16* embbf = (u16*)d_ws;
  u16* wf    = embbf + NV * NE;
  u16* wi    = wf + NH * LDW3;
  u16* wo    = wi + NH * LDW3;
  u16* wc    = wo + NH * LDW3;
  u16* wcls  = wc + NH * LDWC;
  u16* hfin  = wcls + NV * NH;          // [B, H] bf16 final hidden
  size_t bf16_elems = (size_t)(NV * NE + 3 * NH * LDW3 + NH * LDWC + NV * NH
                               + NB * NH);
  size_t bf16_bytes = (bf16_elems * 2 + 255) & ~(size_t)255;
  float* px_c = (float*)((char*)d_ws + bf16_bytes);

  prep_kernel<<<256, 256, 0, stream>>>(emb, W_f, W_i, W_o, W_c, W_cls,
                                       embbf, wf, wi, wo, wc, wcls);

  dim3 pgrid((NT * NB) / 16, 4);
  precompute_kernel<<<pgrid, 256, 0, stream>>>(tokens, embbf, wf, wi, wo, wc,
                                               b_f, b_i, b_o, b_c,
                                               out_f, out_i, out_o, px_c);

  // 16 independent workgroups, one per 16 batch rows — no inter-WGP sync needed
  recurrent_kernel<<<NB / 16, 256, 0, stream>>>(wf, wi, wo, wc,
                                                out_f, out_i, out_o, px_c, hfin);

  classifier_kernel<<<NB / 16, 192, 0, stream>>>(hfin, wcls, b_cls, logits);
}
